// Student_Encoder_Layer_87668872446187
// MI455X (gfx1250) — matrix operand order repaired and checked by reading, not compiled, not run on hardware
//
#include <hip/hip_runtime.h>

// ---------------- problem constants (match reference) ----------------
#define DM   1024           // d_model
#define DFF  4096           // d_ff
#define NB   2              // batch
#define SEQ  2048           // sequence
#define NH   16             // heads
#define HD   64             // head dim
#define ROWS (NB * SEQ)     // 4096 token rows
#define WPITCH 40           // LDS W-tile row pitch in halves (80B) -> conflict-free ds_load_b128

typedef __attribute__((ext_vector_type(16))) _Float16 v16h;
typedef __attribute__((ext_vector_type(8)))  _Float16 v8h;
typedef __attribute__((ext_vector_type(8)))  float    v8f;

static __device__ inline v16h cat8(v8h lo, v8h hi) {
    return __builtin_shufflevector(lo, hi, 0,1,2,3,4,5,6,7,8,9,10,11,12,13,14,15);
}

// A-fragment (16x32 f16, row-major source). Contiguous k order used here (not the ISA table order):
// lane-half h: K = k0 + h*16 + 0..15 (contiguous, matches B)
static __device__ inline v16h load_fragA(const _Float16* __restrict__ row, int k0, int half) {
    const _Float16* p = row + k0 + half * 16;
    return cat8(*(const v8h*)p, *(const v8h*)(p + 8));
}

// A-fragment from an f32 row (attn probs), converted in-register.
static __device__ inline v16h load_fragA_f32(const float* __restrict__ row, int k0, int half) {
    const float* p = row + k0 + half * 16;
    v16h r;
#pragma unroll
    for (int i = 0; i < 8; ++i) { r[i] = (_Float16)p[i]; r[i + 8] = (_Float16)p[i + 8]; }
    return r;
}

// B-fragment (32x16 f16). Column = lane&15; source row is K-contiguous.
// lane-half h holds K = k0 + h*16 + 0..15 (contiguous 32B).
static __device__ inline v16h load_fragB(const _Float16* __restrict__ wrow, int k0, int half) {
    const _Float16* p = wrow + k0 + half * 16;
    return cat8(*(const v8h*)p, *(const v8h*)(p + 8));
}

static __device__ inline v8f wmma_f16(v16h a, v16h b, v8f c) {
    return __builtin_amdgcn_wmma_f32_16x16x32_f16(false, a, false, b, (short)0, c, false, false);
}

// ---- CDNA5 async global->LDS copy (ASYNCcnt-tracked), 16B per lane ----
static __device__ inline void async_copy_b128(unsigned int lds_off, const _Float16* g) {
    asm volatile("global_load_async_to_lds_b128 %0, %1, off"
                 :: "v"(lds_off), "v"((unsigned long long)(uintptr_t)g)
                 : "memory");
}

// ---------------- f32 -> f16 convert ----------------
__global__ void f32_to_f16_kernel(const float* __restrict__ in, _Float16* __restrict__ out, int n) {
    int i = blockIdx.x * blockDim.x + threadIdx.x;
    if (i < n) out[i] = (_Float16)in[i];
}

// ---------------- LayerNorm (f32 in, f16 normalized out) ----------------
__global__ __launch_bounds__(256) void layernorm_kernel(const float* __restrict__ x,
                                                        const float* __restrict__ g,
                                                        const float* __restrict__ bb,
                                                        _Float16* __restrict__ out) {
    const int row = blockIdx.x;
    const int t = threadIdx.x;
    const float* xr = x + (size_t)row * DM;
    __shared__ float s1[256], s2[256];
    float vals[DM / 256];
    float sum = 0.f, sq = 0.f;
#pragma unroll
    for (int i = 0; i < DM / 256; ++i) {
        float v = xr[t + 256 * i];
        vals[i] = v; sum += v; sq += v * v;
    }
    s1[t] = sum; s2[t] = sq; __syncthreads();
    for (int s = 128; s > 0; s >>= 1) {
        if (t < s) { s1[t] += s1[t + s]; s2[t] += s2[t + s]; }
        __syncthreads();
    }
    const float mu   = s1[0] * (1.0f / DM);
    const float var  = s2[0] * (1.0f / DM) - mu * mu;
    const float rstd = rsqrtf(var + 1e-5f);
#pragma unroll
    for (int i = 0; i < DM / 256; ++i) {
        int c = t + 256 * i;
        out[(size_t)row * DM + c] = (_Float16)((vals[i] - mu) * rstd * g[c] + bb[c]);
    }
}

// ---------------- generic WMMA GEMM: out = act(A[M,K] @ W[N,K]^T + bias) ----------------
// W block-tile (128 x 32) staged in LDS via async copies, double buffered.
// OUT mode: 0 = f16 row-major, 1 = f16 scattered into V-transposed [b,h,d,s], 2 = f32 + residual
template <int OUT, bool RELU>
__global__ __launch_bounds__(256) void gemm_wt(const _Float16* __restrict__ A,
                                               const _Float16* __restrict__ W,
                                               const float* __restrict__ bias,
                                               const float* __restrict__ resid,
                                               _Float16* __restrict__ outh,
                                               float* __restrict__ outf,
                                               int M, int N, int K) {
    __shared__ _Float16 wt[2][128][WPITCH];     // 2 x 10KB double buffer
    const int t    = threadIdx.x;
    const int lane = t & 31;
    const int wave = t >> 5;
    const int half = lane >> 4;
    const int l16  = lane & 15;
    const int m0   = blockIdx.y * 64 + (wave >> 1) * 16;
    const int nblk = blockIdx.x * 128;
    const int nw   = (wave & 1) * 64;

    // this thread's two 16B chunks of the 8KB W tile: chunk c -> row c>>2, 16B piece c&3
    const int nl0 = t >> 2,          kc0 = (t & 3) * 8;
    const int nl1 = (t + 256) >> 2,  kc1 = ((t + 256) & 3) * 8;
    const _Float16* wrow0 = W + (size_t)(nblk + nl0) * K + kc0;
    const _Float16* wrow1 = W + (size_t)(nblk + nl1) * K + kc1;

    // prologue: fetch tile for k0 = 0 into buffer 0
    async_copy_b128((unsigned int)(uintptr_t)&wt[0][nl0][kc0], wrow0);
    async_copy_b128((unsigned int)(uintptr_t)&wt[0][nl1][kc1], wrow1);

    v8f acc[4] = {};
    const _Float16* arow = A + (size_t)(m0 + l16) * K;
    int buf = 0;
    for (int k0 = 0; k0 < K; k0 += 32) {
        if (k0 + 32 < K) {
            // prefetch next tile into the other buffer, then retire current tile (in-order loads)
            async_copy_b128((unsigned int)(uintptr_t)&wt[buf ^ 1][nl0][kc0], wrow0 + k0 + 32);
            async_copy_b128((unsigned int)(uintptr_t)&wt[buf ^ 1][nl1][kc1], wrow1 + k0 + 32);
            asm volatile("s_wait_asynccnt 0x2" ::: "memory");
        } else {
            asm volatile("s_wait_asynccnt 0x0" ::: "memory");
        }
        __syncthreads();                       // current tile visible to all waves

        v16h a = load_fragA(arow, k0, half);
#pragma unroll
        for (int s = 0; s < 4; ++s) {
            const _Float16* p = &wt[buf][nw + s * 16 + l16][half * 16];
            v16h b = cat8(*(const v8h*)p, *(const v8h*)(p + 8));
            acc[s] = wmma_f16(a, b, acc[s]);
        }
        __syncthreads();                       // protect buf^1 before next async writes land
        buf ^= 1;
    }

#pragma unroll
    for (int s = 0; s < 4; ++s) {
        const int n = nblk + nw + s * 16 + l16;
        const float bv = bias[n];
#pragma unroll
        for (int i = 0; i < 8; ++i) {
            const int m = m0 + half * 8 + i;
            float v = acc[s][i] + bv;
            if constexpr (RELU) v = fmaxf(v, 0.0f);
            if constexpr (OUT == 0) {
                outh[(size_t)m * N + n] = (_Float16)v;
            } else if constexpr (OUT == 1) {
                // n = h*HD + d ; m = b*SEQ + s  ->  vt[((b*NH+h)*HD + d)*SEQ + s]
                const int hh = n >> 6, dd = n & 63;
                const int bb = m >> 11, ss = m & 2047;
                outh[(((size_t)bb * NH + hh) * HD + dd) * SEQ + ss] = (_Float16)v;
            } else {
                const size_t idx = (size_t)m * N + n;
                outf[idx] = v + resid[idx];
            }
        }
    }
}

// ---------------- fused attention: scores * 1/8, mask, softmax -> probs (f32) ----------------
// One block = one (b,h) and a 16-query strip; 8 waves each own 16 of the 128 key tiles.
__global__ __launch_bounds__(256) void attn_softmax_kernel(const _Float16* __restrict__ Q,
                                                           const _Float16* __restrict__ Kmat,
                                                           const unsigned char* __restrict__ mask,
                                                           float* __restrict__ probs) {
    const int lane = threadIdx.x & 31;
    const int wave = threadIdx.x >> 5;
    const int half = lane >> 4;
    const int l16  = lane & 15;
    const int bh = blockIdx.y, b = bh >> 4, h = bh & 15;
    const int q0 = blockIdx.x * 16;

    const _Float16* Qb = Q    + (size_t)b * SEQ * DM + h * HD;
    const _Float16* Kb = Kmat + (size_t)b * SEQ * DM + h * HD;
    const _Float16* qrow = Qb + (size_t)(q0 + l16) * DM;
    const v16h a0 = load_fragA(qrow, 0,  half);
    const v16h a1 = load_fragA(qrow, 32, half);
    const unsigned char* mrow = mask + (size_t)b * SEQ;

    // scores strip: rows (half*8 + i), cols n_j = (wave + 8*j)*16 + l16
    float sc[16][8];
#pragma unroll
    for (int j = 0; j < 16; ++j) {
        const int n0 = (wave + 8 * j) * 16;
        const _Float16* krow = Kb + (size_t)(n0 + l16) * DM;
        v8f acc = {};
        acc = wmma_f16(a0, load_fragB(krow, 0,  half), acc);
        acc = wmma_f16(a1, load_fragB(krow, 32, half), acc);
        const bool masked = (mrow[n0 + l16] != 0);
#pragma unroll
        for (int i = 0; i < 8; ++i)
            sc[j][i] = masked ? -10000.0f : acc[i] * 0.125f;   // 1/sqrt(64)
    }

    __shared__ float redbuf[8][2][8];   // [wave][half][row i]

    // ---- row max: over 16 tiles (regs), 16 lanes (shfl), 8 waves (LDS) ----
    float rmax[8];
#pragma unroll
    for (int i = 0; i < 8; ++i) {
        float m = sc[0][i];
#pragma unroll
        for (int j = 1; j < 16; ++j) m = fmaxf(m, sc[j][i]);
        for (int d = 1; d < 16; d <<= 1) m = fmaxf(m, __shfl_xor(m, d, 32));
        rmax[i] = m;
    }
    if (l16 == 0) {
#pragma unroll
        for (int i = 0; i < 8; ++i) redbuf[wave][half][i] = rmax[i];
    }
    __syncthreads();
#pragma unroll
    for (int i = 0; i < 8; ++i) {
        float m = redbuf[0][half][i];
#pragma unroll
        for (int w = 1; w < 8; ++w) m = fmaxf(m, redbuf[w][half][i]);
        rmax[i] = m;
    }
    __syncthreads();

    // ---- exp + row sum ----
    float rsum[8];
#pragma unroll
    for (int i = 0; i < 8; ++i) rsum[i] = 0.0f;
#pragma unroll
    for (int j = 0; j < 16; ++j)
#pragma unroll
        for (int i = 0; i < 8; ++i) {
            float e = __expf(sc[j][i] - rmax[i]);
            sc[j][i] = e;
            rsum[i] += e;
        }
#pragma unroll
    for (int i = 0; i < 8; ++i) {
        float s = rsum[i];
        for (int d = 1; d < 16; d <<= 1) s += __shfl_xor(s, d, 32);
        rsum[i] = s;
    }
    if (l16 == 0) {
#pragma unroll
        for (int i = 0; i < 8; ++i) redbuf[wave][half][i] = rsum[i];
    }
    __syncthreads();
    float rinv[8];
#pragma unroll
    for (int i = 0; i < 8; ++i) {
        float s = redbuf[0][half][i];
#pragma unroll
        for (int w = 1; w < 8; ++w) s += redbuf[w][half][i];
        rinv[i] = 1.0f / s;
    }

    // ---- single write of normalized probs ----
    float* sbase = probs + ((size_t)bh * SEQ + q0) * SEQ;
#pragma unroll
    for (int j = 0; j < 16; ++j) {
        const int n = (wave + 8 * j) * 16 + l16;
#pragma unroll
        for (int i = 0; i < 8; ++i) {
            const int row = half * 8 + i;
            sbase[(size_t)row * SEQ + n] = sc[j][i] * rinv[i];
        }
    }
}

// ---------------- attn_out = probs(f32) @ V (V pre-transposed [b,h,d,s]) ----------------
__global__ __launch_bounds__(256) void attn_pv_kernel(const float* __restrict__ P,
                                                      const _Float16* __restrict__ Vt,
                                                      _Float16* __restrict__ ctx) {
    const int lane = threadIdx.x & 31;
    const int wave = threadIdx.x >> 5;
    const int half = lane >> 4;
    const int l16  = lane & 15;
    const int bh = blockIdx.y, b = bh >> 4, h = bh & 15;
    const int m0 = blockIdx.x * 32 + (wave >> 2) * 16;
    const int n0 = (wave & 3) * 16;

    const float*    prow  = P  + ((size_t)bh * SEQ + (m0 + l16)) * SEQ;
    const _Float16* vtrow = Vt + ((size_t)bh * HD + n0 + l16) * SEQ;   // row = head dim d, contiguous in k

    v8f acc = {};
    for (int k0 = 0; k0 < SEQ; k0 += 32) {
        v16h a  = load_fragA_f32(prow, k0, half);
        v16h bf = load_fragB(vtrow, k0, half);
        acc = wmma_f16(a, bf, acc);
    }
    const int n = h * HD + n0 + l16;
#pragma unroll
    for (int i = 0; i < 8; ++i) {
        const int m = m0 + half * 8 + i;
        ctx[((size_t)b * SEQ + m) * DM + n] = (_Float16)acc[i];
    }
}

// ---------------- host launch ----------------
extern "C" void kernel_launch(void* const* d_in, const int* in_sizes, int n_in,
                              void* d_out, int out_size, void* d_ws, size_t ws_size,
                              hipStream_t stream) {
    (void)in_sizes; (void)n_in; (void)out_size; (void)ws_size;
    const float* src  = (const float*)d_in[0];
    const unsigned char* mask = (const unsigned char*)d_in[1]; // bool array (1 byte/elem)
    const float* Wq = (const float*)d_in[2];  const float* bq = (const float*)d_in[3];
    const float* Wk = (const float*)d_in[4];  const float* bk = (const float*)d_in[5];
    const float* Wv = (const float*)d_in[6];  const float* bv = (const float*)d_in[7];
    const float* Wo = (const float*)d_in[8];  const float* bo = (const float*)d_in[9];
    const float* W1 = (const float*)d_in[10]; const float* b1 = (const float*)d_in[11];
    const float* W2 = (const float*)d_in[12]; const float* b2 = (const float*)d_in[13];
    const float* ln1g = (const float*)d_in[14]; const float* ln1b = (const float*)d_in[15];
    const float* ln2g = (const float*)d_in[16]; const float* ln2b = (const float*)d_in[17];

    float* out_src = (float*)d_out;                        // [2,2048,1024] f32
    float* probs   = out_src + (size_t)ROWS * DM;          // [2,16,2048,2048] f32

    // workspace carve-up (f16 buffers, 256B aligned)
    char* ws = (char*)d_ws;
    auto take = [&](size_t elems) {
        _Float16* p = (_Float16*)ws;
        ws += ((elems * sizeof(_Float16)) + 255) & ~(size_t)255;
        return p;
    };
    _Float16* Wq_h = take((size_t)DM * DM);
    _Float16* Wk_h = take((size_t)DM * DM);
    _Float16* Wv_h = take((size_t)DM * DM);
    _Float16* Wo_h = take((size_t)DM * DM);
    _Float16* W1_h = take((size_t)DFF * DM);
    _Float16* W2_h = take((size_t)DM * DFF);
    _Float16* xn1  = take((size_t)ROWS * DM);
    _Float16* xn2  = take((size_t)ROWS * DM);
    _Float16* Qh   = take((size_t)ROWS * DM);
    _Float16* Kh   = take((size_t)ROWS * DM);
    _Float16* Vt   = take((size_t)ROWS * DM);   // V in [b,h,d,s] layout
    _Float16* ctx  = take((size_t)ROWS * DM);
    _Float16* hbuf = take((size_t)ROWS * DFF);

    auto cvt = [&](const float* in, _Float16* out, int n) {
        f32_to_f16_kernel<<<(n + 255) / 256, 256, 0, stream>>>(in, out, n);
    };
    cvt(Wq, Wq_h, DM * DM);
    cvt(Wk, Wk_h, DM * DM);
    cvt(Wv, Wv_h, DM * DM);
    cvt(Wo, Wo_h, DM * DM);
    cvt(W1, W1_h, DFF * DM);
    cvt(W2, W2_h, DM * DFF);

    // LN1
    layernorm_kernel<<<ROWS, 256, 0, stream>>>(src, ln1g, ln1b, xn1);

    // Q, K row-major; V written directly transposed to [b,h,d,s]
    dim3 gproj(DM / 128, ROWS / 64);
    gemm_wt<0, false><<<gproj, 256, 0, stream>>>(xn1, Wq_h, bq, nullptr, Qh, nullptr, ROWS, DM, DM);
    gemm_wt<0, false><<<gproj, 256, 0, stream>>>(xn1, Wk_h, bk, nullptr, Kh, nullptr, ROWS, DM, DM);
    gemm_wt<1, false><<<gproj, 256, 0, stream>>>(xn1, Wv_h, bv, nullptr, Vt, nullptr, ROWS, DM, DM);

    // fused scores + mask + softmax: one pass, one write of the probs tensor
    attn_softmax_kernel<<<dim3(SEQ / 16, NB * NH), 256, 0, stream>>>(Qh, Kh, mask, probs);

    // context = probs @ V  (contiguous B-fragments from Vt)
    attn_pv_kernel<<<dim3(SEQ / 32, NB * NH), 256, 0, stream>>>(probs, Vt, ctx);

    // O projection + residual -> d_out src region
    gemm_wt<2, false><<<gproj, 256, 0, stream>>>(ctx, Wo_h, bo, src, nullptr, out_src, ROWS, DM, DM);

    // LN2, FFN1 (ReLU), FFN2 + residual (overwrites d_out src region)
    layernorm_kernel<<<ROWS, 256, 0, stream>>>(out_src, ln2g, ln2b, xn2);
    gemm_wt<0, true><<<dim3(DFF / 128, ROWS / 64), 256, 0, stream>>>(xn2, W1_h, b1, nullptr, hbuf, nullptr, ROWS, DFF, DM);
    gemm_wt<2, false><<<gproj, 256, 0, stream>>>(hbuf, W2_h, b2, out_src, nullptr, out_src, ROWS, DM, DFF);
}
